// PronunciationModel_76811195122334
// MI455X (gfx1250) — matrix operand order repaired and checked by reading, not compiled, not run on hardware
//
#include <hip/hip_runtime.h>

#define B_  32
#define T_  2000
#define F_  80
#define H_  256
#define S_  100
#define G_  1024   // 4*H
#define D2_ 512    // 2*H

typedef __attribute__((ext_vector_type(16))) __bf16 v16bf;
typedef __attribute__((ext_vector_type(8)))  float  v8f;
typedef unsigned short u16;

// ---- bf16 helpers -----------------------------------------------------------
__device__ inline u16 f2bf_bits(float f) {
    return (u16)((__float_as_uint(f) + 0x8000u) >> 16);
}
__device__ inline unsigned bfpack(float f0, float f1) {
    unsigned u0 = __float_as_uint(f0) + 0x8000u;
    unsigned u1 = __float_as_uint(f1) + 0x8000u;
    return __builtin_amdgcn_perm(u1, u0, 0x07060302u);  // {hi16(u1), hi16(u0)}
}
__device__ inline v16bf mk16(uint4 a, uint4 b) {
    v16bf r;
    __builtin_memcpy(&r, &a, 16);
    __builtin_memcpy((char*)&r + 16, &b, 16);
    return r;
}
// 16 bf16 from two 16B chunks (b128 loads, global or LDS)
__device__ inline v16bf ld_bf16_pair(const u16* p0, const u16* p1) {
    return mk16(*(const uint4*)p0, *(const uint4*)p1);
}
__device__ inline v8f v8zero() {
    v8f r;
#pragma unroll
    for (int i = 0; i < 8; ++i) r[i] = 0.f;
    return r;
}
// fast activations: v_exp_f32 + v_rcp_f32, no IEEE division ladder
__device__ inline float frcp(float x)  { return __builtin_amdgcn_rcpf(x); }
__device__ inline float sigm(float x)  { return frcp(1.f + __expf(-x)); }
__device__ inline float ftanh(float x) { return 1.f - 2.f * frcp(__expf(2.f * x) + 1.f); }

// One 16x16 tile, unrolled K-loop, bf16 A loader (laf(k) -> uint4 = 8 bf16 at
// element offset k), bf16 B row. A frag runs: [kc+8hi,+8) and [kc+16+8hi,+8);
// B frag runs: same as A.
template <int K, typename LAF>
__device__ inline v8f wmma_tile(LAF laf, const u16* wrow, v8f acc, bool hi) {
    const int ao = hi ? 8 : 0, bo = hi ? 16 : 0;
#pragma unroll
    for (int kc = 0; kc < K; kc += 32) {
        v16bf af = mk16(laf(kc + ao), laf(kc + ao + 16));
        v16bf bf = ld_bf16_pair(wrow + kc + ao, wrow + kc + ao + 16);
        acc = __builtin_amdgcn_wmma_f32_16x16x32_bf16(
            false, af, false, bf, (short)0, acc, false, false);
    }
    return acc;
}

// ---- one-time conversions ---------------------------------------------------
__global__ void cvt_bf16_kernel(const float* __restrict__ src,
                                u16* __restrict__ dst, int n2) {
    int i = blockIdx.x * blockDim.x + threadIdx.x;
    if (i >= n2) return;
    ((unsigned*)dst)[i] = bfpack(src[2 * i], src[2 * i + 1]);
}

// conv weights (256,CIN,3) -> bf16 (256,Kpad), k = dt*CIN + c, zero-padded
template <int CIN>
__global__ void conv_wt_kernel(const float* __restrict__ Wc,
                               u16* __restrict__ Wt, int Kpad) {
    int idx = blockIdx.x * blockDim.x + threadIdx.x;
    if (idx >= 256 * Kpad) return;
    int n = idx / Kpad, k = idx - n * Kpad;
    float v = 0.f;
    if (k < 3 * CIN) {
        int dt = k / CIN, c = k - dt * CIN;
        v = Wc[((size_t)n * CIN + c) * 3 + dt];
    }
    Wt[idx] = f2bf_bits(v);
}

// ---- Conv1d(k=3,pad=1)+ReLU as implicit GEMM, bf16 in / bf16 out ------------
template <int CIN, int KPAD>
__global__ void __launch_bounds__(512)
conv_gemm_kernel(const u16* __restrict__ X, const u16* __restrict__ Wt,
                 const float* __restrict__ bias, u16* __restrict__ Y) {
    int w = threadIdx.x >> 5, lane = threadIdx.x & 31;
    bool hi = lane >= 16; int lm = lane & 15;
    int m  = blockIdx.x * 16 + lm;
    int b  = m / T_;
    int tp = m - b * T_;
    int n  = w * 16 + lm;
    const u16* xb = X + (size_t)b * T_ * CIN;
    const int row_off = (tp - 1) * CIN;                   // may be negative
    const int kLo = (tp == 0) ? CIN : 0;
    const int kHi = (tp == T_ - 1) ? 2 * CIN : 3 * CIN;
    const int maxoff = T_ * CIN - 8;
    const u16* wrow = Wt + (size_t)n * KPAD;
    v8f acc; float bv = bias[n];
#pragma unroll
    for (int e = 0; e < 8; ++e) acc[e] = bv;
    acc = wmma_tile<KPAD>(
        [&](int k) -> uint4 {
            int off = row_off + k;
            int oc  = off < 0 ? 0 : (off > maxoff ? maxoff : off);
            uint4 v = *(const uint4*)(xb + oc);
            bool ok = (k >= kLo) & (k + 8 <= kHi);        // 8-elem run uniform (CIN%8==0)
            v.x = ok ? v.x : 0u; v.y = ok ? v.y : 0u;
            v.z = ok ? v.z : 0u; v.w = ok ? v.w : 0u;
            return v;
        },
        wrow, acc, hi);
#pragma unroll
    for (int e = 0; e < 8; ++e) {
        int r = blockIdx.x * 16 + e + (hi ? 8 : 0);
        float v = acc[e];
        Y[(size_t)r * 256 + n] = f2bf_bits(v > 0.f ? v : 0.f);
    }
}

// ---- Generic GEMM: C(f32) = Abf @ Wbf^T + bias (opt ReLU); grid.z slices ----
template <int K>
__global__ void __launch_bounds__(512)
gemm_bias_kernel(const u16* __restrict__ A, int lda,
                 const u16* __restrict__ W, long long wz,
                 const float* __restrict__ bias, long long bz,
                 float* __restrict__ C, long long cz, int ldc, int relu) {
    int w = threadIdx.x >> 5, lane = threadIdx.x & 31;
    bool hi = lane >= 16; int lm = lane & 15;
    const u16*   Wd = W    + (long long)blockIdx.z * wz;
    const float* bd = bias + (long long)blockIdx.z * bz;
    float*       Cd = C    + (long long)blockIdx.z * cz;
    int mBase = blockIdx.x * 16;
    int nBase = blockIdx.y * 256 + w * 16;
    const u16* arow = A  + (size_t)(mBase + lm) * lda;
    const u16* wrow = Wd + (size_t)(nBase + lm) * K;
    v8f acc; float bv = bd[nBase + lm];
#pragma unroll
    for (int e = 0; e < 8; ++e) acc[e] = bv;
    acc = wmma_tile<K>(
        [&](int k) { return *(const uint4*)(arow + k); },
        wrow, acc, hi);
#pragma unroll
    for (int e = 0; e < 8; ++e) {
        int r = mBase + e + (hi ? 8 : 0);
        float v = acc[e];
        if (relu) v = v > 0.f ? v : 0.f;
        Cd[(size_t)r * ldc + nBase + lm] = v;
    }
}

// ---- Sequential BiLSTM: h in LDS (bf16), cell state in VGPRs ----------------
// xproj: (2,B,T,4H) f32; whh_bf: (2,4H,H); yf (f32) / yb (bf16): (B,T,2H)
// 4 blocks = dir*2 + rowTile; 16 waves; wave = 16 hidden cols x all 4 gates.
__global__ void __launch_bounds__(512)
lstm_dir_kernel(const float* __restrict__ xproj, const u16* __restrict__ whh_bf,
                const int* __restrict__ lens, float* __restrict__ yf,
                u16* __restrict__ yb, int wf32, int wbf) {
    __shared__ __align__(16) u16 hb[16][264];
    int dir     = blockIdx.x >> 1;
    int rowBase = (blockIdx.x & 1) * 16;
    int w = threadIdx.x >> 5, lane = threadIdx.x & 31;
    bool hi = lane >= 16; int lm = lane & 15;
    int jcol   = w * 16 + lm;
    int colOff = dir * H_;
    const int ao = hi ? 8 : 0, bo = hi ? 16 : 0;
    const u16*   whhD = whh_bf + (size_t)dir * G_ * H_;
    const float* xpD  = xproj  + (size_t)dir * B_ * T_ * G_;

    for (int i = threadIdx.x; i < 16 * 264 / 2; i += blockDim.x)
        ((unsigned*)hb)[i] = 0u;
    int len8[8];
#pragma unroll
    for (int e = 0; e < 8; ++e) len8[e] = lens[rowBase + e + (hi ? 8 : 0)];
    v8f cstate = v8zero();
    __syncthreads();

    const u16* hr = &hb[lm][0];
    for (int t = 0; t < T_; ++t) {
        v8f acc[4];
#pragma unroll
        for (int e = 0; e < 8; ++e) {
            int b = rowBase + e + (hi ? 8 : 0);
            int L = len8[e];
            int st = (dir == 0) ? t : ((t < L) ? (L - 1 - t) : t);
            const float* xr = xpD + ((size_t)b * T_ + st) * G_;
            acc[0][e] = xr[0 * H_ + jcol];
            acc[1][e] = xr[1 * H_ + jcol];
            acc[2][e] = xr[2 * H_ + jcol];
            acc[3][e] = xr[3 * H_ + jcol];
            // prefetch next step's gather rows (global_prefetch_b8)
            int tn = (t + 1 < T_) ? t + 1 : t;
            int sn = (dir == 0) ? tn : ((tn < L) ? (L - 1 - tn) : tn);
            const float* xp2 = xpD + ((size_t)b * T_ + sn) * G_;
            __builtin_prefetch(xp2 + 0 * H_ + jcol, 0, 1);
            __builtin_prefetch(xp2 + 1 * H_ + jcol, 0, 1);
            __builtin_prefetch(xp2 + 2 * H_ + jcol, 0, 1);
            __builtin_prefetch(xp2 + 3 * H_ + jcol, 0, 1);
        }
#pragma unroll 2
        for (int kc = 0; kc < H_; kc += 32) {
            v16bf af = ld_bf16_pair(hr + kc + ao, hr + kc + ao + 16);
#pragma unroll
            for (int g = 0; g < 4; ++g) {
                const u16* wr = whhD + (size_t)(g * H_ + jcol) * H_ + kc + ao;
                v16bf bf = ld_bf16_pair(wr, wr + 16);
                acc[g] = __builtin_amdgcn_wmma_f32_16x16x32_bf16(
                    false, af, false, bf, (short)0, acc[g], false, false);
            }
        }
        u16 hnew[8];
#pragma unroll
        for (int e = 0; e < 8; ++e) {
            float ig = sigm(acc[0][e]);
            float fg = sigm(acc[1][e]);
            float gg = ftanh(acc[2][e]);
            float og = sigm(acc[3][e]);
            float cn = fg * cstate[e] + ig * gg;
            float hn = og * ftanh(cn);
            int rl = e + (hi ? 8 : 0);
            int L  = len8[e];
            bool ok = t < L;
            u16 hold = hb[rl][jcol];
            cstate[e] = ok ? cn : cstate[e];
            hnew[e] = ok ? f2bf_bits(hn) : hold;
            int b  = rowBase + rl;
            int st = (dir == 0) ? t : ((t < L) ? (L - 1 - t) : t);
            size_t oidx = ((size_t)b * T_ + st) * D2_ + colOff + jcol;
            float yv = ok ? hn : 0.f;
            if (wf32) yf[oidx] = yv;
            if (wbf)  yb[oidx] = ok ? hnew[e] : (u16)0;
        }
        __syncthreads();
#pragma unroll
        for (int e = 0; e < 8; ++e) hb[e + (hi ? 8 : 0)][jcol] = hnew[e];
        __syncthreads();
    }
}

// ---- pooling / segment means (f32 in, bf16 out for head GEMMs) --------------
__global__ void pool_kernel(const float* __restrict__ x, const int* __restrict__ lens,
                            u16* __restrict__ out) {
    int idx = blockIdx.x * blockDim.x + threadIdx.x;
    if (idx >= B_ * D2_) return;
    int b = idx / D2_, d = idx - b * D2_;
    float s = 0.f;
    const float* p = x + (size_t)b * T_ * D2_ + d;
    for (int t = 0; t < T_; ++t) s += p[(size_t)t * D2_];
    int L = lens[b]; if (L < 1) L = 1;
    out[idx] = f2bf_bits(s / (float)L);
}

__global__ void seg_mean_kernel(const float* __restrict__ x,
                                const int* __restrict__ segs,
                                const int* __restrict__ lens,
                                u16* __restrict__ out) {
    int idx = blockIdx.x * blockDim.x + threadIdx.x;
    if (idx >= B_ * S_ * D2_) return;
    int b = idx / (S_ * D2_);
    int r = idx - b * (S_ * D2_);
    int s = r / D2_, d = r - s * D2_;
    int L  = lens[b];
    int s0 = segs[(b * S_ + s) * 2 + 0];
    int e0 = segs[(b * S_ + s) * 2 + 1];
    if (s0 < 0) s0 = 0; if (s0 > L) s0 = L;
    if (e0 > L) e0 = L; if (e0 < s0 + 1) e0 = s0 + 1;
    float sum = 0.f;
    const float* p = x + (size_t)b * T_ * D2_ + d;
    for (int t = s0; t < e0; ++t) sum += p[(size_t)t * D2_];
    out[idx] = f2bf_bits(sum / (float)(e0 - s0));
}

__global__ void utt_out_kernel(const float* __restrict__ hid,
                               const float* __restrict__ w2,
                               const float* __restrict__ b2,
                               float* __restrict__ out) {
    int b = threadIdx.x;
    if (b >= B_) return;
    float s = b2[0];
    for (int d = 0; d < D2_; ++d) s += hid[(size_t)b * D2_ + d] * w2[d];
    out[b] = s;
}

__global__ void seg_out_kernel(const float* __restrict__ hid,
                               const float* __restrict__ w2,
                               const float* __restrict__ b2,
                               const int* __restrict__ seg_lens,
                               float* __restrict__ out) {
    int idx = blockIdx.x * blockDim.x + threadIdx.x;
    if (idx >= B_ * S_) return;
    int b = idx / S_, s = idx - b * S_;
    float v = b2[0];
    for (int d = 0; d < D2_; ++d) v += hid[(size_t)idx * D2_ + d] * w2[d];
    out[idx] = (s < seg_lens[b]) ? v : 0.f;
}

// ---- launch -----------------------------------------------------------------
extern "C" void kernel_launch(void* const* d_in, const int* in_sizes, int n_in,
                              void* d_out, int out_size, void* d_ws, size_t ws_size,
                              hipStream_t stream) {
    const float* features = (const float*)d_in[0];
    const int*   flens    = (const int*)d_in[1];
    const int*   segs     = (const int*)d_in[2];
    const int*   slens    = (const int*)d_in[3];
    const float* c1w = (const float*)d_in[4];
    const float* c1b = (const float*)d_in[5];
    const float* c2w = (const float*)d_in[6];
    const float* c2b = (const float*)d_in[7];
    const float* wih0 = (const float*)d_in[8];
    const float* whh0 = (const float*)d_in[9];
    const float* b0   = (const float*)d_in[10];
    const float* wih1 = (const float*)d_in[11];
    const float* whh1 = (const float*)d_in[12];
    const float* b1   = (const float*)d_in[13];
    const float* segw1 = (const float*)d_in[14];
    const float* segb1 = (const float*)d_in[15];
    const float* segw2 = (const float*)d_in[16];
    const float* segb2 = (const float*)d_in[17];
    const float* uttw1 = (const float*)d_in[18];
    const float* uttb1 = (const float*)d_in[19];
    const float* uttw2 = (const float*)d_in[20];
    const float* uttb2 = (const float*)d_in[21];
    float* out = (float*)d_out;

    const size_t n_bt = (size_t)B_ * T_;
    // ---- u16 (bf16) region --------------------------------------------------
    u16* wb = (u16*)d_ws;
    u16* wt1     = wb;                       // 256*256
    u16* wt2     = wt1 + 256 * 256;          // 256*768
    u16* wih0b   = wt2 + 256 * 768;          // 2*1024*256
    u16* whh0b   = wih0b + 2 * G_ * 256;     // 2*1024*256
    u16* wih1b   = whh0b + 2 * G_ * 256;     // 2*1024*512
    u16* whh1b   = wih1b + 2 * G_ * 512;     // 2*1024*256
    u16* segw1b  = whh1b + 2 * G_ * 256;     // 512*512
    u16* uttw1b  = segw1b + D2_ * D2_;       // 512*512
    u16* feat_bf = uttw1b + D2_ * D2_;       // B*T*80
    u16* conv1_bf = feat_bf + n_bt * F_;     // B*T*256
    u16* conv2_bf = conv1_bf + n_bt * 256;   // B*T*256
    u16* lstm0_bf = conv2_bf + n_bt * 256;   // B*T*512
    u16* lstm1_bf = lstm0_bf + n_bt * D2_;   // B*T*512 (unused sink)
    u16* pooled_bf = lstm1_bf + n_bt * D2_;  // B*512
    u16* segmean_bf = pooled_bf + B_ * D2_;  // B*S*512
    u16* wend = segmean_bf + (size_t)B_ * S_ * D2_;
    // ---- f32 region ---------------------------------------------------------
    float* fbp = (float*)wend;               // offset is 16B-multiple
    float* xproj     = fbp;                              // 2*B*T*1024 (reused)
    float* lstm1_out = xproj + 2 * n_bt * (size_t)G_;    // B*T*512
    float* utt_hid   = lstm1_out + n_bt * D2_;           // B*512
    float* seg_hid   = utt_hid + (size_t)B_ * D2_;       // B*S*512

    const int mtiles = (B_ * T_) / 16;                   // 4000
    const int TB = 256;
    auto cvt = [&](const float* s, u16* d, size_t n) {
        cvt_bf16_kernel<<<dim3((unsigned)((n / 2 + TB - 1) / TB)), TB, 0, stream>>>(s, d, (int)(n / 2));
    };

    // one-time conversions
    cvt(features, feat_bf, n_bt * F_);
    cvt(wih0, wih0b, 2 * G_ * 256);
    cvt(whh0, whh0b, 2 * G_ * 256);
    cvt(wih1, wih1b, 2 * G_ * 512);
    cvt(whh1, whh1b, 2 * G_ * 256);
    cvt(segw1, segw1b, D2_ * D2_);
    cvt(uttw1, uttw1b, D2_ * D2_);
    conv_wt_kernel<F_><<<dim3(256 * 256 / TB), TB, 0, stream>>>(c1w, wt1, 256);
    conv_wt_kernel<256><<<dim3(256 * 768 / TB), TB, 0, stream>>>(c2w, wt2, 768);

    // conv stack (bf16 in, bf16 out)
    conv_gemm_kernel<F_, 256><<<dim3(mtiles), 512, 0, stream>>>(feat_bf, wt1, c1b, conv1_bf);
    conv_gemm_kernel<256, 768><<<dim3(mtiles), 512, 0, stream>>>(conv1_bf, wt2, c2b, conv2_bf);

    // layer 0
    gemm_bias_kernel<256><<<dim3(mtiles, 4, 2), 512, 0, stream>>>(
        conv2_bf, 256, wih0b, (long long)G_ * 256, b0, G_,
        xproj, (long long)n_bt * G_, G_, 0);
    lstm_dir_kernel<<<dim3(4), 512, 0, stream>>>(
        xproj, whh0b, flens, (float*)nullptr, lstm0_bf, 0, 1);

    // layer 1
    gemm_bias_kernel<512><<<dim3(mtiles, 4, 2), 512, 0, stream>>>(
        lstm0_bf, D2_, wih1b, (long long)G_ * D2_, b1, G_,
        xproj, (long long)n_bt * G_, G_, 0);
    lstm_dir_kernel<<<dim3(4), 512, 0, stream>>>(
        xproj, whh1b, flens, lstm1_out, lstm1_bf, 1, 0);

    // pooling + heads
    pool_kernel<<<dim3((B_ * D2_ + TB - 1) / TB), TB, 0, stream>>>(lstm1_out, flens, pooled_bf);
    seg_mean_kernel<<<dim3((B_ * S_ * D2_ + TB - 1) / TB), TB, 0, stream>>>(
        lstm1_out, segs, flens, segmean_bf);

    gemm_bias_kernel<512><<<dim3(B_ / 16, D2_ / 256, 1), 512, 0, stream>>>(
        pooled_bf, D2_, uttw1b, 0, uttb1, 0, utt_hid, 0, D2_, 1);
    gemm_bias_kernel<512><<<dim3((B_ * S_) / 16, D2_ / 256, 1), 512, 0, stream>>>(
        segmean_bf, D2_, segw1b, 0, segb1, 0, seg_hid, 0, D2_, 1);

    utt_out_kernel<<<dim3(1), 32, 0, stream>>>(utt_hid, uttw2, uttb2, out);
    seg_out_kernel<<<dim3((B_ * S_ + TB - 1) / TB), TB, 0, stream>>>(
        seg_hid, segw2, segb2, slens, out + B_);
}
